// PianoMuseRWKV_43791486550516
// MI455X (gfx1250) — matrix operand order repaired and checked by reading, not compiled, not run on hardware
//
#include <hip/hip_runtime.h>
#include <hip/hip_bf16.h>

// ---------- types (ext_vector: trivial, union-safe, match builtin operands) ----------
typedef __bf16 bf16_t;
typedef __attribute__((ext_vector_type(16))) __bf16 bf16x16;
typedef __attribute__((ext_vector_type(8)))  float  f32x8;
typedef __attribute__((ext_vector_type(4)))  int    i32x4;
typedef __attribute__((ext_vector_type(4)))  float  f32x4;

__device__ __forceinline__ unsigned short bfbits(float f) {
    union { float f; unsigned u; } c; c.f = f;
    unsigned r = c.u + 0x7FFFu + ((c.u >> 16) & 1u);   // round-to-nearest-even
    return (unsigned short)(r >> 16);
}
__device__ __forceinline__ bf16_t f2bf(float f) {
    union { unsigned short s; bf16_t b; } o; o.s = bfbits(f);
    return o.b;
}
__device__ __forceinline__ unsigned pk2(float lo, float hi) {   // bf16(lo) | bf16(hi)<<16
    return (unsigned)bfbits(lo) | ((unsigned)bfbits(hi) << 16);
}

// ---------------------------------------------------------------------------
// GEMM: C[M,N] (f32) = A[M,K] (bf16 row-major) x W[K,N] (f32 row-major, cvt->bf16)
// block = 256 thr = 8 waves; block tile 64x128; wave tile 32x32; K-step 32.
// Per wave per K-step: 4 x v_wmma_f32_16x16x32_bf16 (2 A-frags x 2 B-frags).
// K, N compile-time -> constant strides (shift/add addressing, immediate offsets).
// ALIGNED=true -> N-tile fully in range, no bounds checks anywhere.
// ---------------------------------------------------------------------------
#define TM 64
#define TN 128
#define TK 32

template<int K, int N, bool ALIGNED>
__global__ __launch_bounds__(256) void gemm_bf16_wmma(
    const bf16_t* __restrict__ A, const float* __restrict__ W,
    float* __restrict__ C, int colOffset)
{
    __shared__ __attribute__((aligned(16))) bf16_t Blds[TN * TK]; // [n][k] transposed, 8KB

    const int tid  = threadIdx.x;
    const int lane = tid & 31;
    const int wave = tid >> 5;
    const int tm   = wave & 1;       // 2 waves along M (32 rows each)
    const int tn   = wave >> 1;      // 4 waves along N (32 cols each)
    const int rowBase = blockIdx.x * TM;
    const int colBase = colOffset + blockIdx.y * TN;

    f32x8 acc00 = {0.f,0.f,0.f,0.f,0.f,0.f,0.f,0.f};
    f32x8 acc01 = acc00, acc10 = acc00, acc11 = acc00;

    const int half = lane >> 4;                            // K-half selector
    const int m0   = rowBase + tm * 32 + (lane & 15);      // A rows for this lane
    const bf16_t* __restrict__ arow0 = A + (size_t)m0 * K;
    const bf16_t* __restrict__ arow1 = arow0 + (size_t)16 * K;

    union Frag { i32x4 i[2]; bf16x16 v; };
    unsigned* __restrict__ bl32 = (unsigned*)&Blds[0];

    for (int k0 = 0; k0 < K; k0 += TK) {
        // ---- stage W[k0..k0+31, colBase..+127] -> Blds[n][k] bf16, packed b32 stores ----
        #pragma unroll
        for (int r = 0; r < 2; ++r) {
            int p  = tid + r * 256;          // 0..511 : one 2(K)x4(N) patch each
            int kp = (p >> 5) << 1;          // 0,2,...,30
            int n4 = (p & 31) << 2;          // 0..124
            int gn = colBase + n4;
            const float* wp0 = W + (size_t)(k0 + kp) * N + gn;
            const float* wp1 = wp0 + N;
            f32x4 w0, w1;
            if (ALIGNED || gn + 3 < N) {
                w0 = *(const f32x4*)wp0;
                w1 = *(const f32x4*)wp1;
            } else {
                w0.x = (gn+0<N)?wp0[0]:0.f; w0.y = (gn+1<N)?wp0[1]:0.f;
                w0.z = (gn+2<N)?wp0[2]:0.f; w0.w = (gn+3<N)?wp0[3]:0.f;
                w1.x = (gn+0<N)?wp1[0]:0.f; w1.y = (gn+1<N)?wp1[1]:0.f;
                w1.z = (gn+2<N)?wp1[2]:0.f; w1.w = (gn+3<N)?wp1[3]:0.f;
            }
            const int kd = kp >> 1;          // dword index along K
            bl32[(n4 + 0) * (TK/2) + kd] = pk2(w0.x, w1.x);
            bl32[(n4 + 1) * (TK/2) + kd] = pk2(w0.y, w1.y);
            bl32[(n4 + 2) * (TK/2) + kd] = pk2(w0.z, w1.z);
            bl32[(n4 + 3) * (TK/2) + kd] = pk2(w0.w, w1.w);
        }
        __syncthreads();

        // ---- A fragments: lanes 0-15 K{0..7,16..23}, lanes 16-31 K{8..15,24..31} ----
        Frag a0f, a1f;
        a0f.i[0] = *(const i32x4*)(arow0 + k0 + half * 8);
        a0f.i[1] = *(const i32x4*)(arow0 + k0 + 16 + half * 8);
        a1f.i[0] = *(const i32x4*)(arow1 + k0 + half * 8);
        a1f.i[1] = *(const i32x4*)(arow1 + k0 + 16 + half * 8);

        // ---- B fragments from LDS: lane holds col (lane&15), K{half*8..+7, 16+half*8..+7} ----
        const int kb = half * 8;
        const bf16_t* bp0 = &Blds[(tn * 32 + (lane & 15)) * TK + kb];
        const bf16_t* bp1 = bp0 + 16 * TK;
        Frag b0, b1;
        b0.i[0] = *(const i32x4*)(bp0);
        b0.i[1] = *(const i32x4*)(bp0 + 16);
        b1.i[0] = *(const i32x4*)(bp1);
        b1.i[1] = *(const i32x4*)(bp1 + 16);

        acc00 = __builtin_amdgcn_wmma_f32_16x16x32_bf16(false, a0f.v, false, b0.v,
                                                        (short)0, acc00, false, false);
        acc01 = __builtin_amdgcn_wmma_f32_16x16x32_bf16(false, a0f.v, false, b1.v,
                                                        (short)0, acc01, false, false);
        acc10 = __builtin_amdgcn_wmma_f32_16x16x32_bf16(false, a1f.v, false, b0.v,
                                                        (short)0, acc10, false, false);
        acc11 = __builtin_amdgcn_wmma_f32_16x16x32_bf16(false, a1f.v, false, b1.v,
                                                        (short)0, acc11, false, false);
        __syncthreads();
    }

    // ---- store: VGPR r -> M = half*8 + r within 16-row subtile; ptr-increment ----
    const int nn0 = colBase + tn * 32 + (lane & 15);
    const int mr0 = rowBase + tm * 32 + half * 8;
    float* __restrict__ c0 = C + (size_t)mr0 * N + nn0;        // rows mr0..mr0+7
    float* __restrict__ c1 = c0 + (size_t)16 * N;              // rows mr0+16..+23
    #pragma unroll
    for (int r = 0; r < 8; ++r) {
        if (ALIGNED || nn0 < N)      { c0[0]  = acc00[r]; c1[0]  = acc10[r]; }
        if (ALIGNED || nn0 + 16 < N) { c0[16] = acc01[r]; c1[16] = acc11[r]; }
        c0 += N; c1 += N;
    }
}

// ---------------------------------------------------------------------------
// Elementwise / LN / WKV kernels
// ---------------------------------------------------------------------------
__global__ __launch_bounds__(256) void gather_emb(
    const int* __restrict__ ids, const float* __restrict__ emb,
    float* __restrict__ out, int D)
{
    int row = blockIdx.x;
    const float* e = emb + (size_t)ids[row] * D;
    float* o = out + (size_t)row * D;
    for (int d = threadIdx.x; d < D; d += 256) o[d] = e[d];
}

// LayerNorm; optionally also writes bf16 copy (out16 may be null).
__global__ __launch_bounds__(256) void ln_kernel(
    const float* __restrict__ in, const float* __restrict__ w,
    const float* __restrict__ b, float* __restrict__ out,
    bf16_t* __restrict__ out16, int D)
{
    const int row = blockIdx.x;
    const float* x = in + (size_t)row * D;
    float* o = out + (size_t)row * D;
    float s = 0.f, s2 = 0.f;
    for (int d = threadIdx.x; d < D; d += 256) { float v = x[d]; s += v; s2 += v * v; }
    __shared__ float sh[256], sh2[256];
    sh[threadIdx.x] = s; sh2[threadIdx.x] = s2;
    __syncthreads();
    for (int st = 128; st > 0; st >>= 1) {
        if (threadIdx.x < st) { sh[threadIdx.x] += sh[threadIdx.x + st];
                                sh2[threadIdx.x] += sh2[threadIdx.x + st]; }
        __syncthreads();
    }
    float mean = sh[0] / D;
    float var  = sh2[0] / D - mean * mean;
    float rstd = rsqrtf(var + 1e-5f);
    for (int d = threadIdx.x; d < D; d += 256) {
        float v = (x[d] - mean) * rstd * w[d] + b[d];
        o[d] = v;
        if (out16) out16[(size_t)row * D + d] = f2bf(v);
    }
}

// out = h*mix + shift(h)*(1-mix); ov may be null (channel-mix has only k,r)
__global__ __launch_bounds__(256) void shift_mix(
    const float* __restrict__ h,
    const float* __restrict__ mk, const float* __restrict__ mv, const float* __restrict__ mr,
    bf16_t* __restrict__ ok, bf16_t* __restrict__ ov, bf16_t* __restrict__ orr,
    int T, int D, int total)
{
    int gid = blockIdx.x * 256 + threadIdx.x;
    if (gid >= total) return;
    int d = gid % D;
    int n = gid / D;
    int t = n % T;                       // n = b*T + t
    float hv = h[gid];
    float hp = (t == 0) ? 0.f : h[gid - D];
    float k = mk[d]; ok[gid]  = f2bf(hv * k + hp * (1.f - k));
    if (ov) { float v = mv[d]; ov[gid] = f2bf(hv * v + hp * (1.f - v)); }
    float r = mr[d]; orr[gid] = f2bf(hv * r + hp * (1.f - r));
}

// RWKV-4 numerically-stable scan; one lane per (b,d) channel; emits sigmoid(r)*wkv.
__global__ __launch_bounds__(256) void wkv_gate(
    const float* __restrict__ kk, const float* __restrict__ vv,
    const float* __restrict__ rr,
    const float* __restrict__ td, const float* __restrict__ tf,
    bf16_t* __restrict__ gated, int Bn, int T, int D)
{
    int c = blockIdx.x * 256 + threadIdx.x;
    if (c >= Bn * D) return;
    int d = c % D, b = c / D;
    float w = -__expf(td[d]);
    float u = tf[d];
    float aa = 0.f, bb = 0.f, pp = -1e38f;
    size_t idx = (size_t)b * T * D + d;
    for (int t = 0; t < T; ++t, idx += D) {
        float kt = kk[idx], vt = vv[idx];
        float ww = u + kt;
        float p  = fmaxf(pp, ww);
        float e1 = __expf(pp - p), e2 = __expf(ww - p);
        float o  = (e1 * aa + e2 * vt) / (e1 * bb + e2);
        float sg = 1.f / (1.f + __expf(-rr[idx]));
        gated[idx] = f2bf(sg * o);
        ww = pp + w;
        p  = fmaxf(ww, kt);
        e1 = __expf(ww - p); e2 = __expf(kt - p);
        aa = e1 * aa + e2 * vt;
        bb = e1 * bb + e2;
        pp = p;
    }
}

__global__ __launch_bounds__(256) void add_inplace(float* __restrict__ x,
                                                   const float* __restrict__ y, int n)
{
    int i = blockIdx.x * 256 + threadIdx.x;
    if (i < n) x[i] += y[i];
}

__global__ __launch_bounds__(256) void sigmul_add(float* __restrict__ x,
                                                  const float* __restrict__ cr,
                                                  const float* __restrict__ cv, int n)
{
    int i = blockIdx.x * 256 + threadIdx.x;
    if (i < n) x[i] += (1.f / (1.f + __expf(-cr[i]))) * cv[i];
}

__global__ __launch_bounds__(256) void relu2_bf16(const float* __restrict__ in,
                                                  bf16_t* __restrict__ out, int n)
{
    int i = blockIdx.x * 256 + threadIdx.x;
    if (i < n) { float v = fmaxf(in[i], 0.f); out[i] = f2bf(v * v); }
}

// ---------------------------------------------------------------------------
extern "C" void kernel_launch(void* const* d_in, const int* in_sizes, int n_in,
                              void* d_out, int out_size, void* d_ws, size_t ws_size,
                              hipStream_t stream)
{
    (void)in_sizes; (void)n_in; (void)out_size; (void)ws_size;
    constexpr int L = 12, D = 768, F = 3072, V = 50257, B = 2, T = 1024;
    constexpr int N = B * T;

    const int*   ids   = (const int*)  d_in[0];
    const float* emb   = (const float*)d_in[1];
    const float* ln0w  = (const float*)d_in[2];
    const float* ln0b  = (const float*)d_in[3];
    const float* ln1w  = (const float*)d_in[4];
    const float* ln1b  = (const float*)d_in[5];
    const float* ln2w  = (const float*)d_in[6];
    const float* ln2b  = (const float*)d_in[7];
    const float* td    = (const float*)d_in[8];
    const float* tf    = (const float*)d_in[9];
    const float* mixk  = (const float*)d_in[10];
    const float* mixv  = (const float*)d_in[11];
    const float* mixr  = (const float*)d_in[12];
    const float* Wk    = (const float*)d_in[13];
    const float* Wv    = (const float*)d_in[14];
    const float* Wr    = (const float*)d_in[15];
    const float* Wo    = (const float*)d_in[16];
    const float* cmk   = (const float*)d_in[17];
    const float* cmr   = (const float*)d_in[18];
    const float* cWk   = (const float*)d_in[19];
    const float* cWv   = (const float*)d_in[20];
    const float* cWr   = (const float*)d_in[21];
    const float* lnow  = (const float*)d_in[22];
    const float* lnob  = (const float*)d_in[23];
    const float* headW = (const float*)d_in[24];
    float* out = (float*)d_out;

    // workspace carve-up (~78 MB)
    char* ws = (char*)d_ws;
    float*  x    = (float*)ws;  ws += (size_t)N * D * 4;
    float*  h    = (float*)ws;  ws += (size_t)N * D * 4;
    float*  f0   = (float*)ws;  ws += (size_t)N * D * 4;   // kk / cr / ao
    float*  f1   = (float*)ws;  ws += (size_t)N * D * 4;   // vv / cv
    float*  f2   = (float*)ws;  ws += (size_t)N * D * 4;   // rr
    float*  kff  = (float*)ws;  ws += (size_t)N * F * 4;   // FFN pre-activation
    bf16_t* a0   = (bf16_t*)ws; ws += (size_t)N * D * 2;   // xk16 / h16
    bf16_t* a1   = (bf16_t*)ws; ws += (size_t)N * D * 2;   // xv16
    bf16_t* a2   = (bf16_t*)ws; ws += (size_t)N * D * 2;   // xr16 / gated
    bf16_t* kf16 = (bf16_t*)ws; ws += (size_t)N * F * 2;

    const int ND = N * D, NF = N * F;
    auto cdiv = [](int a, int b) { return (a + b - 1) / b; };
    dim3 gD(N / TM, D / TN);                 // 32 x 6
    dim3 gF(N / TM, F / TN);                 // 32 x 24
    dim3 gVa(N / TM, V / TN);                // 32 x 392 (aligned bulk of head)
    dim3 gVe(N / TM, 1);                     // ragged last 81 columns

    gather_emb<<<N, 256, 0, stream>>>(ids, emb, h, D);
    ln_kernel <<<N, 256, 0, stream>>>(h, ln0w, ln0b, x, nullptr, D);

    for (int i = 0; i < L; ++i) {
        // ---- time mixing ----
        ln_kernel<<<N, 256, 0, stream>>>(x, ln1w + i * D, ln1b + i * D, h, nullptr, D);
        shift_mix<<<cdiv(ND, 256), 256, 0, stream>>>(h, mixk + i * D, mixv + i * D,
                                                     mixr + i * D, a0, a1, a2, T, D, ND);
        gemm_bf16_wmma<D, D, true><<<gD, 256, 0, stream>>>(a0, Wk + (size_t)i * D * D, f0, 0);
        gemm_bf16_wmma<D, D, true><<<gD, 256, 0, stream>>>(a1, Wv + (size_t)i * D * D, f1, 0);
        gemm_bf16_wmma<D, D, true><<<gD, 256, 0, stream>>>(a2, Wr + (size_t)i * D * D, f2, 0);
        wkv_gate<<<cdiv(B * D, 256), 256, 0, stream>>>(f0, f1, f2, td + i * D, tf + i * D,
                                                       a2, B, T, D);
        gemm_bf16_wmma<D, D, true><<<gD, 256, 0, stream>>>(a2, Wo + (size_t)i * D * D, f0, 0);
        add_inplace<<<cdiv(ND, 256), 256, 0, stream>>>(x, f0, ND);
        // ---- channel mixing ----
        ln_kernel<<<N, 256, 0, stream>>>(x, ln2w + i * D, ln2b + i * D, h, nullptr, D);
        shift_mix<<<cdiv(ND, 256), 256, 0, stream>>>(h, cmk + i * D, nullptr,
                                                     cmr + i * D, a0, nullptr, a2, T, D, ND);
        gemm_bf16_wmma<D, F, true><<<gF, 256, 0, stream>>>(a0, cWk + (size_t)i * D * F, kff, 0);
        relu2_bf16<<<cdiv(NF, 256), 256, 0, stream>>>(kff, kf16, NF);
        gemm_bf16_wmma<D, D, true><<<gD, 256, 0, stream>>>(a2, cWr + (size_t)i * D * D, f0, 0);
        gemm_bf16_wmma<F, D, true><<<gD, 256, 0, stream>>>(kf16, cWv + (size_t)i * F * D, f1, 0);
        sigmul_add<<<cdiv(ND, 256), 256, 0, stream>>>(x, f0, f1, ND);
    }

    // ---- head: aligned bulk + ragged tail column-block ----
    ln_kernel<<<N, 256, 0, stream>>>(x, lnow, lnob, h, a0, D);
    gemm_bf16_wmma<D, V, true ><<<gVa, 256, 0, stream>>>(a0, headW, out, 0);
    gemm_bf16_wmma<D, V, false><<<gVe, 256, 0, stream>>>(a0, headW, out, (V / TN) * TN);
}
